// Votenet_90752658964837
// MI455X (gfx1250) — compile-verified, matrix operand order checked by reading, not run on hardware
//
#include <hip/hip_runtime.h>

// ---------------------------------------------------------------------------
// VoteNet forward for gfx1250 (MI455X). MLP (1x1 conv) layers run on
// v_wmma_f32_16x16x32_f16 with zero-padded operands so the inner loop has
// NO guards: 4x global_load_b128 (f32 A) + 2x 16B loads (f16 packed W),
// 16x v_cvt, 1x wmma per 32-deep k-step.
// ---------------------------------------------------------------------------

typedef __attribute__((ext_vector_type(16))) _Float16 v16h;
typedef __attribute__((ext_vector_type(8)))  _Float16 v8h;
typedef __attribute__((ext_vector_type(8)))  float    v8f;

#define BB 8
#define NN 20000

// ---------------------------------------------------------------------------
// Weight pack: f32 [cout, cin] -> f16 zero-padded [CoutP, Kp]
// ---------------------------------------------------------------------------
__global__ void k_packW(const float* __restrict__ W, _Float16* __restrict__ Wh,
                        int cin, int cout, int Kp, long total)
{
    long t = (long)blockIdx.x * blockDim.x + threadIdx.x;
    if (t >= total) return;
    int k = (int)(t % Kp);
    int c = (int)(t / Kp);
    float v = (c < cout && k < cin) ? W[(long)c * cin + k] : 0.0f;
    Wh[t] = (_Float16)v;
}

// ---------------------------------------------------------------------------
// WMMA GEMM:  Y[M, CoutP] = X[M, Kp] * Wh[CoutP, Kp]^T + bias
// X rows are Kp-strided (Kp % 32 == 0), pad region zero.  Wh pre-packed f16.
// One wave computes one 16x16 tile; guard-free inner loop.
// ---------------------------------------------------------------------------
__global__ void k_wmma_gemm(const float* __restrict__ X,
                            const _Float16* __restrict__ Wh,
                            const float* __restrict__ bias,
                            float* __restrict__ Y,
                            int Kp, int Cout, int CoutP, int MT, int NT)
{
    int wave = blockIdx.x * (blockDim.x >> 5) + (threadIdx.x >> 5);
    if (wave >= MT * NT) return;                 // wave-uniform guard
    int lane = threadIdx.x & 31;
    int tm = wave % MT;
    int tn = wave / MT;
    int hsel = lane >> 4;
    int r    = lane & 15;

    const float*    xrow = X  + (size_t)(tm * 16 + r) * Kp;
    const _Float16* wrow = Wh + (size_t)(tn * 16 + r) * Kp;

    v8f acc = {};
    for (int k0 = 0; k0 < Kp; k0 += 32) {
        // ISA 16-bit A layout: lanes 0-15 hold K {k0..k0+7, k0+16..k0+23},
        // lanes 16-31 hold K {k0+8..k0+15, k0+24..k0+31}.
        int base = k0 + hsel * 8;
        const float4* pa = (const float4*)(xrow + base);        // 16B aligned
        float4 a0 = pa[0];
        float4 a1 = pa[1];
        const float4* pa2 = (const float4*)(xrow + base + 16);
        float4 a2 = pa2[0];
        float4 a3 = pa2[1];
        v8h b0 = *(const v8h*)(wrow + base);                    // 16B aligned
        v8h b1 = *(const v8h*)(wrow + base + 16);

        v16h a, b;
        a[0]  = (_Float16)a0.x; a[1]  = (_Float16)a0.y;
        a[2]  = (_Float16)a0.z; a[3]  = (_Float16)a0.w;
        a[4]  = (_Float16)a1.x; a[5]  = (_Float16)a1.y;
        a[6]  = (_Float16)a1.z; a[7]  = (_Float16)a1.w;
        a[8]  = (_Float16)a2.x; a[9]  = (_Float16)a2.y;
        a[10] = (_Float16)a2.z; a[11] = (_Float16)a2.w;
        a[12] = (_Float16)a3.x; a[13] = (_Float16)a3.y;
        a[14] = (_Float16)a3.z; a[15] = (_Float16)a3.w;
#pragma unroll
        for (int e = 0; e < 8; ++e) { b[e] = b0[e]; b[8 + e] = b1[e]; }

        acc = __builtin_amdgcn_wmma_f32_16x16x32_f16(
            false, a, false, b, (short)0, acc, false, false);
    }

    int col = tn * 16 + r;
    bool cok = (col < Cout);
    float bb = cok ? bias[col] : 0.0f;
#pragma unroll
    for (int j = 0; j < 8; ++j) {
        int rr = tm * 16 + j + 8 * hsel;
        Y[(size_t)rr * CoutP + col] = cok ? (acc[j] + bb) : 0.0f;  // keep pad zero
    }
}

// ---------------------------------------------------------------------------
// BatchNorm statistics over all M rows (stride Cp), one block per real channel.
// ---------------------------------------------------------------------------
__global__ void k_bn_stats(const float* __restrict__ Y,
                           float* __restrict__ mean, float* __restrict__ msq,
                           int M, int Cp)
{
    int c = blockIdx.x;
    float s = 0.f, s2 = 0.f;
    for (int i = threadIdx.x; i < M; i += blockDim.x) {
        float v = Y[(long)i * Cp + c];
        s += v; s2 += v * v;
    }
    __shared__ float sh[256], sh2[256];
    sh[threadIdx.x] = s; sh2[threadIdx.x] = s2;
    __syncthreads();
    for (int o = blockDim.x >> 1; o > 0; o >>= 1) {
        if ((int)threadIdx.x < o) { sh[threadIdx.x] += sh[threadIdx.x + o];
                                    sh2[threadIdx.x] += sh2[threadIdx.x + o]; }
        __syncthreads();
    }
    if (threadIdx.x == 0) { mean[c] = sh[0] / (float)M; msq[c] = sh2[0] / (float)M; }
}

__global__ void k_bn_relu(float* __restrict__ Y,
                          const float* __restrict__ mean, const float* __restrict__ msq,
                          const float* __restrict__ g, const float* __restrict__ be,
                          long total, int C, int Cp)
{
    long t = (long)blockIdx.x * blockDim.x + threadIdx.x;
    if (t >= total) return;                      // total = M*Cp
    int c = (int)(t % Cp);
    if (c >= C) return;                          // pad columns stay zero
    float m   = mean[c];
    float var = msq[c] - m * m;
    float y = g[c] * (Y[t] - m) * rsqrtf(var + 1e-5f) + be[c];
    Y[t] = y > 0.f ? y : 0.f;
}

// ---------------------------------------------------------------------------
// Farthest point sampling: one block per batch.
// ---------------------------------------------------------------------------
__global__ void k_fps(const float* __restrict__ xyz, float* __restrict__ dist,
                      int* __restrict__ outIdx, int N, int S)
{
    int b = blockIdx.x;
    const float* p = xyz + (long)b * N * 3;
    float* d = dist + (long)b * N;
    __shared__ float smax[512];
    __shared__ int   simx[512];
    __shared__ int   sfar;
    int tid = threadIdx.x;
    for (int i = tid; i < N; i += blockDim.x) d[i] = 1e10f;
    if (tid == 0) sfar = 0;
    __syncthreads();
    for (int s = 0; s < S; ++s) {
        int far = sfar;
        if (tid == 0) outIdx[b * S + s] = far;
        float cx = p[far * 3 + 0], cy = p[far * 3 + 1], cz = p[far * 3 + 2];
        float best = -1.f; int bi = 0;
        for (int i = tid; i < N; i += blockDim.x) {
            float dx = p[i * 3 + 0] - cx;
            float dy = p[i * 3 + 1] - cy;
            float dz = p[i * 3 + 2] - cz;
            float dd = fminf(d[i], dx * dx + dy * dy + dz * dz);
            d[i] = dd;
            if (dd > best) { best = dd; bi = i; }
        }
        smax[tid] = best; simx[tid] = bi;
        __syncthreads();
        for (int o = blockDim.x >> 1; o > 0; o >>= 1) {
            if (tid < o && smax[tid + o] > smax[tid]) {
                smax[tid] = smax[tid + o]; simx[tid] = simx[tid + o];
            }
            __syncthreads();
        }
        if (tid == 0) sfar = simx[0];
        __syncthreads();
    }
}

__global__ void k_gather(const float* __restrict__ in, const int* __restrict__ idx,
                         float* __restrict__ out, int N, int S, int C, long total)
{
    long t = (long)blockIdx.x * blockDim.x + threadIdx.x;
    if (t >= total) return;
    int c = (int)(t % C);
    long bs = t / C;
    int s = (int)(bs % S);
    int b = (int)(bs / S);
    int j = idx[b * S + s];
    out[t] = in[((long)b * N + j) * C + c];
}

__global__ void k_gather_i(const int* __restrict__ in, const int* __restrict__ idx,
                           int* __restrict__ out, int N, int S, long total)
{
    long t = (long)blockIdx.x * blockDim.x + threadIdx.x;
    if (t >= total) return;
    int b = (int)(t / S);
    out[t] = in[(long)b * N + idx[t]];
}

// ball query: first K in-radius neighbors (ascending index), pad with first.
__global__ void k_ball(const float* __restrict__ xyz, const float* __restrict__ nxyz,
                       int* __restrict__ gidx, int N, int S, int K, float r2, long total)
{
    long t = (long)blockIdx.x * blockDim.x + threadIdx.x;
    if (t >= total) return;
    int s = (int)(t % S);
    int b = (int)(t / S);
    const float* p = xyz + (long)b * N * 3;
    const float* c = nxyz + ((long)b * S + s) * 3;
    float cx = c[0], cy = c[1], cz = c[2];
    int* g = gidx + t * K;
    int cnt = 0;
    for (int j = 0; j < N && cnt < K; ++j) {
        float dx = p[j * 3 + 0] - cx;
        float dy = p[j * 3 + 1] - cy;
        float dz = p[j * 3 + 2] - cz;
        if (dx * dx + dy * dy + dz * dz <= r2) g[cnt++] = j;
    }
    int f = (cnt > 0) ? g[0] : 0;
    for (; cnt < K; ++cnt) g[cnt] = f;
}

// grouped tensor [B,S,K, CinP]: centered xyz ++ features, zero pad columns
__global__ void k_group(const float* __restrict__ xyz, const float* __restrict__ nxyz,
                        const float* __restrict__ feats, const int* __restrict__ gidx,
                        float* __restrict__ out, int N, int S, int K, int Cf, int CinP,
                        long total)
{
    long t = (long)blockIdx.x * blockDim.x + threadIdx.x;
    if (t >= total) return;                       // total = B*S*K
    long bs = t / K;
    int s = (int)(bs % S);
    int b = (int)(bs / S);
    int j = gidx[t];
    float* o = out + t * CinP;
    const float* pj = xyz + ((long)b * N + j) * 3;
    const float* c  = nxyz + ((long)b * S + s) * 3;
    o[0] = pj[0] - c[0];
    o[1] = pj[1] - c[1];
    o[2] = pj[2] - c[2];
    if (Cf > 0) {
        const float* f = feats + ((long)b * N + j) * Cf;
        for (int cc = 0; cc < Cf; ++cc) o[3 + cc] = f[cc];
    }
    for (int cc = 3 + Cf; cc < CinP; ++cc) o[cc] = 0.0f;
}

// max over nsample axis: F[B,S,K, stride Cs] -> out[B,S,C]
__global__ void k_maxpool(const float* __restrict__ F, float* __restrict__ out,
                          int S, int K, int C, int Cs, long total)
{
    long t = (long)blockIdx.x * blockDim.x + threadIdx.x;
    if (t >= total) return;                       // total = B*S*C
    int c = (int)(t % C);
    long bs = t / C;
    const float* f = F + (bs * K) * Cs + c;
    float m = -1e30f;
    for (int k = 0; k < K; ++k) m = fmaxf(m, f[(long)k * Cs]);
    out[t] = m;
}

// 3-NN inverse-distance interpolation of points2 onto xyz1
__global__ void k_interp3(const float* __restrict__ xyz1, const float* __restrict__ xyz2,
                          const float* __restrict__ p2, float* __restrict__ out,
                          int N1, int N2, int C2, long total)
{
    long t = (long)blockIdx.x * blockDim.x + threadIdx.x;
    if (t >= total) return;                       // total = B*N1
    int i = (int)(t % N1);
    int b = (int)(t / N1);
    const float* q = xyz1 + ((long)b * N1 + i) * 3;
    const float* p = xyz2 + (long)b * N2 * 3;
    float qx = q[0], qy = q[1], qz = q[2];
    float d0 = 1e30f, d1 = 1e30f, d2 = 1e30f;
    int   i0 = 0,     i1 = 0,     i2 = 0;
    for (int j = 0; j < N2; ++j) {
        float dx = p[j * 3 + 0] - qx;
        float dy = p[j * 3 + 1] - qy;
        float dz = p[j * 3 + 2] - qz;
        float dd = dx * dx + dy * dy + dz * dz;
        if (dd < d0)      { d2 = d1; i2 = i1; d1 = d0; i1 = i0; d0 = dd; i0 = j; }
        else if (dd < d1) { d2 = d1; i2 = i1; d1 = dd; i1 = j; }
        else if (dd < d2) { d2 = dd; i2 = j; }
    }
    float w0 = 1.f / (d0 + 1e-8f), w1 = 1.f / (d1 + 1e-8f), w2 = 1.f / (d2 + 1e-8f);
    float ws = w0 + w1 + w2;
    w0 /= ws; w1 /= ws; w2 /= ws;
    const float* f0 = p2 + ((long)b * N2 + i0) * C2;
    const float* f1 = p2 + ((long)b * N2 + i1) * C2;
    const float* f2 = p2 + ((long)b * N2 + i2) * C2;
    float* o = out + ((long)b * N1 + i) * C2;
    for (int c = 0; c < C2; ++c) o[c] = w0 * f0[c] + w1 * f1[c] + w2 * f2[c];
}

// channel concat into padded rows: out[.., Cp] = [A | B | 0pad]
__global__ void k_concat(const float* __restrict__ A, int C1,
                         const float* __restrict__ Bf, int C2,
                         float* __restrict__ out, int Cp, long rows)
{
    long total = rows * Cp;
    long t = (long)blockIdx.x * blockDim.x + threadIdx.x;
    if (t >= total) return;
    long row = t / Cp;
    int c = (int)(t % Cp);
    float v = 0.0f;
    if (c < C1) v = A[row * C1 + c];
    else if (c < C1 + C2) v = Bf[row * C2 + (c - C1)];
    out[t] = v;
}

// vote split (net stride netS): xyz += net[0:3], feat += net[3:259]
__global__ void k_vote(const float* __restrict__ l1x, const float* __restrict__ l1p,
                       const float* __restrict__ net, int netS,
                       float* __restrict__ vxyz, float* __restrict__ vfeat, long total)
{
    long t = (long)blockIdx.x * blockDim.x + threadIdx.x;
    if (t >= total) return;                       // total = B*256*259
    int ch = (int)(t % 259);
    long bs = t / 259;
    float nv = net[bs * netS + ch];
    if (ch < 3) vxyz[bs * 3 + ch] = l1x[bs * 3 + ch] + nv;
    else        vfeat[bs * 256 + (ch - 3)] = l1p[bs * 256 + (ch - 3)] + nv;
}

// proposal head (h stride hS): out[...,1:4] += agg_xyz
__global__ void k_proposal(const float* __restrict__ h, int hS,
                           const float* __restrict__ axyz,
                           float* __restrict__ o8, long total)
{
    long t = (long)blockIdx.x * blockDim.x + threadIdx.x;
    if (t >= total) return;                       // total = B*128*8
    int ch = (int)(t % 8);
    long bq = t / 8;
    float v = h[bq * hS + ch];
    if (ch >= 1 && ch < 4) v += axyz[bq * 3 + (ch - 1)];
    o8[t] = v;
}

// (B,S,C) -> (B,C,S)
__global__ void k_transpose(const float* __restrict__ in, float* __restrict__ out,
                            int S, int C, long total)
{
    long t = (long)blockIdx.x * blockDim.x + threadIdx.x;
    if (t >= total) return;
    int c = (int)(t % C);
    long bs = t / C;
    int s = (int)(bs % S);
    int b = (int)(bs / S);
    out[((long)b * C + c) * S + s] = in[t];
}

__global__ void k_copy(const float* __restrict__ in, float* __restrict__ out, long total)
{
    long t = (long)blockIdx.x * blockDim.x + threadIdx.x;
    if (t < total) out[t] = in[t];
}

__global__ void k_seedout(const int* __restrict__ seed1, const int* __restrict__ fip,
                          float* __restrict__ out, long total)
{
    long t = (long)blockIdx.x * blockDim.x + threadIdx.x;
    if (t >= total) return;                       // total = B*128
    int b = (int)(t / 128);
    out[t] = (float)seed1[(long)b * 256 + fip[t]];
}

// ---------------------------------------------------------------------------
// Host orchestration
// ---------------------------------------------------------------------------
struct L4 { const float *W, *b, *g, *be; _Float16* Wh; int cin, cout, kp, coutP; };

static inline int pad32(int x) { return (x + 31) & ~31; }

extern "C" void kernel_launch(void* const* d_in, const int* in_sizes, int n_in,
                              void* d_out, int out_size, void* d_ws, size_t ws_size,
                              hipStream_t stream)
{
    (void)in_sizes; (void)n_in; (void)out_size; (void)ws_size;

    const float* xyz       = (const float*)d_in[0];
    const int*   seed_inds = (const int*)  d_in[1];

    int pi = 2;
    auto P = [&]() -> const float* { return (const float*)d_in[pi++]; };
    auto mkbn = [&](int ci, int co) { L4 l; l.W = P(); l.b = P(); l.g = P(); l.be = P();
                                      l.Wh = nullptr; l.cin = ci; l.cout = co;
                                      l.kp = pad32(ci); l.coutP = pad32(co); return l; };
    auto mknb = [&](int ci, int co) { L4 l; l.W = P(); l.b = P(); l.g = nullptr; l.be = nullptr;
                                      l.Wh = nullptr; l.cin = ci; l.cout = co;
                                      l.kp = pad32(ci); l.coutP = pad32(co); return l; };

    L4 sa1[3]; sa1[0] = mkbn(3, 64);    sa1[1] = mkbn(64, 64);   sa1[2] = mkbn(64, 128);
    L4 sa2[3]; sa2[0] = mkbn(131, 128); sa2[1] = mkbn(128, 128); sa2[2] = mkbn(128, 256);
    L4 sa3[3]; sa3[0] = mkbn(259, 128); sa3[1] = mkbn(128, 128); sa3[2] = mkbn(128, 256);
    L4 sa4[3]; sa4[0] = mkbn(259, 128); sa4[1] = mkbn(128, 128); sa4[2] = mkbn(128, 256);
    L4 fp4[2]; fp4[0] = mkbn(512, 256); fp4[1] = mkbn(256, 256);
    L4 fp3[2]; fp3[0] = mkbn(512, 256); fp3[1] = mkbn(256, 256);
    L4 fp2[2]; fp2[0] = mkbn(384, 256); fp2[1] = mkbn(256, 256);
    L4 vg1 = mkbn(256, 256), vg2 = mkbn(256, 256), vg3 = mknb(256, 259);
    L4 psa[4]; psa[0] = mkbn(259, 259); psa[1] = mkbn(259, 128); psa[2] = mkbn(128, 128); psa[3] = mkbn(128, 128);
    L4 pn1 = mkbn(128, 128), pn2 = mkbn(128, 128), pn3 = mknb(128, 8);

    // ---- workspace carve ---------------------------------------------------
    char* w = (char*)d_ws; size_t off = 0;
    auto A  = [&](size_t n) -> float*    { float* p = (float*)(w + off);    off += ((n * 4 + 255) & ~(size_t)255); return p; };
    auto AI = [&](size_t n) -> int*      { int*   p = (int*)  (w + off);    off += ((n * 4 + 255) & ~(size_t)255); return p; };
    auto AH = [&](size_t n) -> _Float16* { _Float16* p = (_Float16*)(w + off); off += ((n * 2 + 255) & ~(size_t)255); return p; };

    const size_t BIGN = (size_t)65536 * 288;      // largest padded activation tensor
    float* BIG0 = A(BIGN);
    float* BIG1 = A(BIGN);
    float* l1x  = A((size_t)BB * 256 * 3);  float* l1p   = A((size_t)BB * 256 * 128);
    float* l2x  = A((size_t)BB * 128 * 3);  float* l2p   = A((size_t)BB * 128 * 256);
    float* l3x  = A((size_t)BB * 64 * 3);   float* l3p   = A((size_t)BB * 64 * 256);
    float* l4x  = A((size_t)BB * 64 * 3);   float* l4p   = A((size_t)BB * 64 * 256);
    float* l3pn = A((size_t)BB * 64 * 256);
    float* l2pn = A((size_t)BB * 128 * 256);
    float* l1pn = A((size_t)BB * 256 * 256);
    float* netb = A((size_t)BB * 256 * 288);      // vg3 output, stride 288
    float* vxyz = A((size_t)BB * 256 * 3);  float* vfeat = A((size_t)BB * 256 * 256);
    float* axyz = A((size_t)BB * 128 * 3);  float* afeat = A((size_t)BB * 128 * 128);
    float* hbuf = A((size_t)BB * 128 * 32); // pn3 output, stride 32
    float* o8   = A((size_t)BB * 128 * 8);
    float* itp  = A((size_t)BB * 256 * 256);
    float* mean = A(512);                   float* msq   = A(512);
    float* fdst = A((size_t)BB * NN);
    int* fi1  = AI((size_t)BB * 256);
    int* fip  = AI((size_t)BB * 128);
    int* gidx = AI((size_t)BB * 256 * 64);
    int* sed1 = AI((size_t)BB * 256);
    int* tmpi = AI((size_t)BB * 256);

    auto GRD = [](long total) { return dim3((unsigned)((total + 255) / 256)); };

    // pack all weights into padded f16 (deterministic, every call)
    auto prep = [&](L4& l) {
        long tot = (long)l.coutP * l.kp;
        l.Wh = AH((size_t)tot);
        k_packW<<<GRD(tot), dim3(256), 0, stream>>>(l.W, l.Wh, l.cin, l.cout, l.kp, tot);
    };
    for (int i = 0; i < 3; ++i) { prep(sa1[i]); prep(sa2[i]); prep(sa3[i]); prep(sa4[i]); }
    for (int i = 0; i < 2; ++i) { prep(fp4[i]); prep(fp3[i]); prep(fp2[i]); }
    prep(vg1); prep(vg2); prep(vg3);
    for (int i = 0; i < 4; ++i) prep(psa[i]);
    prep(pn1); prep(pn2); prep(pn3);

    auto gemm = [&](const float* X, const L4& l, float* Y, int M) {
        int MT = M / 16, NT = l.coutP / 16;
        int tiles = MT * NT;
        int blocks = (tiles + 7) / 8;             // 8 waves (256 threads) per block
        k_wmma_gemm<<<dim3(blocks), dim3(256), 0, stream>>>(X, l.Wh, l.b, Y, l.kp, l.cout, l.coutP, MT, NT);
    };
    auto bnrelu = [&](float* Y, const L4& l, int M) {
        k_bn_stats<<<dim3(l.cout), dim3(256), 0, stream>>>(Y, mean, msq, M, l.coutP);
        long tot = (long)M * l.coutP;
        k_bn_relu<<<GRD(tot), dim3(256), 0, stream>>>(Y, mean, msq, l.g, l.be, tot, l.cout, l.coutP);
    };
    // MLP chain; input must already be in BIG0 (padded to Ls[0].kp)
    auto run_mlp = [&](int M, const L4* Ls, int n, float* finalOut) -> float* {
        float* X = BIG0;
        for (int i = 0; i < n; ++i) {
            float* Y = (i == n - 1 && finalOut) ? finalOut : ((X == BIG0) ? BIG1 : BIG0);
            gemm(X, Ls[i], Y, M);
            if (Ls[i].g) bnrelu(Y, Ls[i], M);
            X = Y;
        }
        return X;
    };
    auto set_abs = [&](const float* xin, int Nin, const float* pin, int Cf,
                       int S, int K, float radius, L4* Ls, int n,
                       float* oxyz, float* ofeat, int* oidx) {
        k_fps<<<dim3(BB), dim3(512), 0, stream>>>(xin, fdst, oidx, Nin, S);
        long tg = (long)BB * S * 3;
        k_gather<<<GRD(tg), dim3(256), 0, stream>>>(xin, oidx, oxyz, Nin, S, 3, tg);
        long tb = (long)BB * S;
        k_ball<<<GRD(tb), dim3(256), 0, stream>>>(xin, oxyz, gidx, Nin, S, K, radius * radius, tb);
        long tk = (long)BB * S * K;
        k_group<<<GRD(tk), dim3(256), 0, stream>>>(xin, oxyz, pin, gidx, BIG0, Nin, S, K, Cf, Ls[0].kp, tk);
        float* F = run_mlp((int)tk, Ls, n, nullptr);
        int Cl = Ls[n - 1].cout, Cs = Ls[n - 1].coutP;
        long tm = (long)BB * S * Cl;
        k_maxpool<<<GRD(tm), dim3(256), 0, stream>>>(F, ofeat, S, K, Cl, Cs, tm);
    };
    auto fprop = [&](const float* x1, int N1, const float* x2, int N2,
                     const float* p1, int C1, const float* p2, int C2,
                     L4* Ls, int n, float* outp) {
        long ti = (long)BB * N1;
        k_interp3<<<GRD(ti), dim3(256), 0, stream>>>(x1, x2, p2, itp, N1, N2, C2, ti);
        k_concat<<<GRD((long)BB * N1 * Ls[0].kp), dim3(256), 0, stream>>>(p1, C1, itp, C2, BIG0, Ls[0].kp, (long)BB * N1);
        run_mlp(BB * N1, Ls, n, outp);
    };

    // ---- pipeline ----------------------------------------------------------
    set_abs(xyz, NN, nullptr, 0, 256, 64, 0.2f, sa1, 3, l1x, l1p, fi1);
    {
        long ts = (long)BB * 256;
        k_gather_i<<<GRD(ts), dim3(256), 0, stream>>>(seed_inds, fi1, sed1, NN, 256, ts);
    }
    set_abs(l1x, 256, l1p, 128, 128, 64, 0.4f, sa2, 3, l2x, l2p, tmpi);
    set_abs(l2x, 128, l2p, 256, 64, 64, 0.8f, sa3, 3, l3x, l3p, tmpi);
    set_abs(l3x, 64,  l3p, 256, 64, 64, 0.8f, sa4, 3, l4x, l4p, tmpi);

    fprop(l3x, 64,  l4x, 64,  l3p, 256, l4p,  256, fp4, 2, l3pn);
    fprop(l2x, 128, l3x, 64,  l2p, 256, l3pn, 256, fp3, 2, l2pn);
    fprop(l1x, 256, l2x, 128, l1p, 128, l2pn, 256, fp2, 2, l1pn);

    // vote generation (l1pn stride 256 == vg1.kp)
    gemm(l1pn, vg1, BIG1, 2048); bnrelu(BIG1, vg1, 2048);
    gemm(BIG1, vg2, BIG0, 2048); bnrelu(BIG0, vg2, 2048);
    gemm(BIG0, vg3, netb, 2048);                  // netb stride 288
    {
        long tv = (long)BB * 256 * 259;
        k_vote<<<GRD(tv), dim3(256), 0, stream>>>(l1x, l1pn, netb, 288, vxyz, vfeat, tv);
    }

    // proposal set-abstraction on votes
    set_abs(vxyz, 256, vfeat, 256, 128, 64, 0.3f, psa, 4, axyz, afeat, fip);

    // proposal head (afeat stride 128 == pn1.kp)
    gemm(afeat, pn1, BIG1, 1024); bnrelu(BIG1, pn1, 1024);
    gemm(BIG1, pn2, BIG0, 1024); bnrelu(BIG0, pn2, 1024);
    gemm(BIG0, pn3, hbuf, 1024);                  // hbuf stride 32
    {
        long tp = (long)BB * 128 * 8;
        k_proposal<<<GRD(tp), dim3(256), 0, stream>>>(hbuf, 32, axyz, o8, tp);
    }

    // ---- outputs (flattened tuple, all as float) ---------------------------
    float* out = (float*)d_out;
    {
        long t0 = (long)BB * 256 * 3;             // transpose(l1x) -> (B,3,256)
        k_transpose<<<GRD(t0), dim3(256), 0, stream>>>(l1x, out + 0, 256, 3, t0);
        long t1 = (long)BB * 256 * 3;             // vote_xyz (B,256,3)
        k_copy<<<GRD(t1), dim3(256), 0, stream>>>(vxyz, out + 6144, t1);
        long t2 = (long)BB * 128 * 8;             // transpose(out) -> (B,8,128)
        k_transpose<<<GRD(t2), dim3(256), 0, stream>>>(o8, out + 12288, 128, 8, t2);
        long t3 = (long)BB * 128;                 // seed_inds_out (B,128)
        k_seedout<<<GRD(t3), dim3(256), 0, stream>>>(sed1, fip, out + 20480, t3);
    }
}
